// Test_TwoEinsum_62878321214341
// MI455X (gfx1250) — matrix operand order repaired and checked by reading, not compiled, not run on hardware
//
#include <hip/hip_runtime.h>
#include <hip/hip_bf16.h>

typedef __bf16 bf16_t;
typedef __bf16 v16bf __attribute__((ext_vector_type(16)));
typedef __bf16 v8bf  __attribute__((ext_vector_type(8)));
typedef float  v8f   __attribute__((ext_vector_type(8)));
typedef float  v4f   __attribute__((ext_vector_type(4)));
typedef unsigned int u32x4 __attribute__((ext_vector_type(4)));
typedef int          i32x4 __attribute__((ext_vector_type(4)));
typedef int          i32x8 __attribute__((ext_vector_type(8)));

#define BM 128
#define BN 128
#define BK 32
#define LSTRIDE 40  // padded LDS row stride (elements): 32 data + 8 pad

#if __has_builtin(__builtin_amdgcn_tensor_load_to_lds) && \
    __has_builtin(__builtin_amdgcn_s_wait_tensorcnt)
#define USE_TDM 1
#else
#define USE_TDM 0
#endif

// ---------------------------------------------------------------------------
// f32 -> bf16 conversion (layout-preserving), 8 elements/thread.
// ---------------------------------------------------------------------------
__global__ __launch_bounds__(256) void cvt_f32_bf16(const float* __restrict__ src,
                                                    bf16_t* __restrict__ dst,
                                                    long n) {
  long i = ((long)blockIdx.x * blockDim.x + threadIdx.x) * 8;
  if (i + 8 <= n) {
    v4f a = *(const v4f*)(src + i);
    v4f b = *(const v4f*)(src + i + 4);
    v8bf o;
    o[0] = (bf16_t)a[0]; o[1] = (bf16_t)a[1]; o[2] = (bf16_t)a[2]; o[3] = (bf16_t)a[3];
    o[4] = (bf16_t)b[0]; o[5] = (bf16_t)b[1]; o[6] = (bf16_t)b[2]; o[7] = (bf16_t)b[3];
    *(v8bf*)(dst + i) = o;
  }
}

// ---------------------------------------------------------------------------
// f32 -> bf16 conversion WITH transpose (LDS-tiled 32x32), per z-slice:
//   dst[c][r] = (bf16) src[r][c],  src is rows x cols row-major.
// ---------------------------------------------------------------------------
__global__ __launch_bounds__(256) void cvt_tr_f32_bf16(const float* __restrict__ src,
                                                       bf16_t* __restrict__ dst,
                                                       int rows, int cols,
                                                       long srcStrideZ, long dstStrideZ) {
  __shared__ float st[32][33];
  const int tx = threadIdx.x & 31;
  const int ty = threadIdx.x >> 5;
  const long z = blockIdx.z;
  const float* s = src + z * srcStrideZ;
  bf16_t* d = dst + z * dstStrideZ;
  const int r0 = blockIdx.y * 32;  // src row tile
  const int c0 = blockIdx.x * 32;  // src col tile
#pragma unroll
  for (int j = 0; j < 4; j++)
    st[ty + j * 8][tx] = s[(long)(r0 + ty + j * 8) * cols + (c0 + tx)];
  __syncthreads();
#pragma unroll
  for (int j = 0; j < 4; j++)
    d[(long)(c0 + ty + j * 8) * rows + (r0 + tx)] = (bf16_t)st[tx][ty + j * 8];
}

// ---------------------------------------------------------------------------
// Deterministic ordered reduction over 8 expert partials -> bf16.
// ---------------------------------------------------------------------------
__global__ __launch_bounds__(256) void reduce8_bf16(const float* __restrict__ src,
                                                    bf16_t* __restrict__ dst,
                                                    int n, long stride) {
  int i = blockIdx.x * blockDim.x + threadIdx.x;
  if (i < n) {
    float s = 0.0f;
#pragma unroll
    for (int e = 0; e < 8; e++) s += src[(long)e * stride + i];
    dst[i] = (bf16_t)s;
  }
}

#if USE_TDM
// ---------------------------------------------------------------------------
// TDM: DMA a 128-row x 32-element bf16 tile (row stride `ld` elements) from
// global into LDS at byte offset `lds_off`, padding each 64B row with 16B so
// the LDS image lands with stride LSTRIDE=40 elements (bank de-conflicted).
// D# packing per CDNA5 ISA ch.8 (group0 128b, group1 256b; groups 2/3 zero
// for a 2D tile: tile_dim2 = 0). 6-arg builtin variant on this toolchain.
// ---------------------------------------------------------------------------
__device__ __forceinline__ void tdm_load_tile_128x32(const bf16_t* gsrc,
                                                     unsigned lds_off, int ld) {
  unsigned long long ga = (unsigned long long)(size_t)gsrc;
  u32x4 g0;
  g0[0] = 1u;                                   // count=1, user descriptor
  g0[1] = lds_off;                              // lds_addr (bytes)
  g0[2] = (unsigned)ga;                         // global_addr[31:0]
  g0[3] = (unsigned)((ga >> 32) & 0x1ffffffull) // global_addr[56:32]
          | (2u << 30);                         // type=2 ("image")
  const int td0 = ld;                           // tensor_dim0 (OOB bound)
  const int td1 = 1 << 20;                      // tensor_dim1: ample rows
  i32x8 g1;
  g1[0] = (1 << 16)      // data_size = 2 bytes
        | (1 << 20)      // pad_enable
        | (3 << 22)      // pad_interval: every 16 DWORDs (= one 64B tile row)
        | (3 << 25);     // pad_amount: 4 DWORDs (16B) -> LDS stride 80B
  g1[1] = (td0 & 0xffff) << 16;                          // tensor_dim0[15:0]
  g1[2] = ((td0 >> 16) & 0xffff) | ((td1 & 0xffff) << 16);
  g1[3] = ((td1 >> 16) & 0xffff) | (BK << 16);           // tile_dim0 = 32 elems
  g1[4] = BM & 0xffff;                                   // tile_dim1 = 128 rows
  g1[5] = ld;                                            // dim0_stride[31:0]
  g1[6] = 0;                                             // dim0_stride[47:32]
  g1[7] = 0;
  i32x4 gz;
  gz[0] = 0; gz[1] = 0; gz[2] = 0; gz[3] = 0;
  i32x8 gz8;
  gz8[0] = 0; gz8[1] = 0; gz8[2] = 0; gz8[3] = 0;
  gz8[4] = 0; gz8[5] = 0; gz8[6] = 0; gz8[7] = 0;
  __builtin_amdgcn_tensor_load_to_lds(g0, g1, gz, gz, gz8, 0);
}
#endif

// ---------------------------------------------------------------------------
// One 32-deep WMMA step: 4x2 tiles of v_wmma_f32_16x16x32_bf16 per wave.
// Both operands are K-contiguous in LDS (stride LSTRIDE), so every fragment
// is two contiguous ds_load_b128.
// ---------------------------------------------------------------------------
__device__ __forceinline__ void mma_step(const bf16_t* sAbuf, const bf16_t* sBbuf,
                                         int wm0, int wn0, int l16, int lhalf,
                                         v8f acc[4][2]) {
  v16bf afrag[4];
#pragma unroll
  for (int mt = 0; mt < 4; mt++) {
    int m = wm0 + mt * 16 + l16;
    v8bf lo = *(const v8bf*)(sAbuf + m * LSTRIDE + lhalf * 8);
    v8bf hi = *(const v8bf*)(sAbuf + m * LSTRIDE + lhalf * 8 + 16);
    afrag[mt] = __builtin_shufflevector(lo, hi, 0, 1, 2, 3, 4, 5, 6, 7,
                                        8, 9, 10, 11, 12, 13, 14, 15);
  }
  v16bf bfrag[2];
#pragma unroll
  for (int nt = 0; nt < 2; nt++) {
    int n = wn0 + nt * 16 + l16;
    v8bf lo = *(const v8bf*)(sBbuf + n * LSTRIDE + lhalf * 8);
    v8bf hi = *(const v8bf*)(sBbuf + n * LSTRIDE + lhalf * 8 + 16);
    bfrag[nt] = __builtin_shufflevector(lo, hi, 0, 1, 2, 3, 4, 5, 6, 7,
                                        8, 9, 10, 11, 12, 13, 14, 15);
  }
#pragma unroll
  for (int mt = 0; mt < 4; mt++)
#pragma unroll
    for (int nt = 0; nt < 2; nt++)
      acc[mt][nt] = __builtin_amdgcn_wmma_f32_16x16x32_bf16(
          false, afrag[mt], false, bfrag[nt], (short)0, acc[mt][nt], false, false);
}

// ---------------------------------------------------------------------------
// NT-GEMM, bf16 in / f32 accumulate:
//   C[z] = A[z] (M x K, lda) * B[z]^T where B[z] is stored N x K (ldb).
// storeCT != 0 -> store C transposed (contiguous 16B vector stores).
// Staging: TDM double-buffered DMA (TENSORcnt) when available.
// ---------------------------------------------------------------------------
__global__ __launch_bounds__(256) void gemm_bf16_wmma(
    const bf16_t* __restrict__ A, const bf16_t* __restrict__ B,
    float* __restrict__ C,
    int lda, int ldb, int ldc, int K,
    long aStrideZ, long bStrideZ, long cStrideZ, int storeCT) {
  __shared__ __align__(16) bf16_t sA[2][BM * LSTRIDE];
  __shared__ __align__(16) bf16_t sB[2][BN * LSTRIDE];

  const int tid  = threadIdx.x;
  const int lane = tid & 31;
  // wave id as a true SGPR value so the TDM-issue branch is a scalar branch
  const int wv = __builtin_amdgcn_readfirstlane((int)tid) >> 5;

  const long z = blockIdx.z;
  A += z * aStrideZ;
  B += z * bStrideZ;
  C += z * cStrideZ;

  const int m0 = blockIdx.y * BM;
  const int n0 = blockIdx.x * BN;

  const int wave = tid >> 5;
  const int wm0 = (wave >> 2) * 64;  // 2x4 wave grid, 64x32 per wave
  const int wn0 = (wave & 3) * 32;
  const int lhalf = lane >> 4;
  const int l16   = lane & 15;

  const bf16_t* Arow = A + (long)m0 * lda;
  const bf16_t* Brow = B + (long)n0 * ldb;

  v8f acc[4][2];
#pragma unroll
  for (int i = 0; i < 4; i++)
#pragma unroll
    for (int j = 0; j < 2; j++)
#pragma unroll
      for (int q = 0; q < 8; q++) acc[i][j][q] = 0.0f;

  const int nsteps = K / BK;

#if USE_TDM
  if (wv == 0) {
    tdm_load_tile_128x32(Arow, (unsigned)(size_t)&sA[0][0], lda);
    tdm_load_tile_128x32(Brow, (unsigned)(size_t)&sB[0][0], ldb);
  }
  for (int s = 0; s < nsteps; s++) {
    const int buf = s & 1;
    const bool more = (s + 1) < nsteps;
    if (more) {
      if (wv == 0) {
        tdm_load_tile_128x32(Arow + (s + 1) * BK, (unsigned)(size_t)&sA[buf ^ 1][0], lda);
        tdm_load_tile_128x32(Brow + (s + 1) * BK, (unsigned)(size_t)&sB[buf ^ 1][0], ldb);
        // retire the older DMA pair; keep the newer pair (2 ops) in flight
        __builtin_amdgcn_s_wait_tensorcnt((short)2);
      }
    } else {
      if (wv == 0) __builtin_amdgcn_s_wait_tensorcnt((short)0);
    }
    __syncthreads();
    mma_step(&sA[buf][0], &sB[buf][0], wm0, wn0, l16, lhalf, acc);
    __syncthreads();  // everyone done reading buf before it is re-targeted
  }
#else
  // Fallback: synchronous vector-copy staging, single-buffered.
  const int ac0 = tid * 2;  // 512 chunks of 8 elems per 128x32 tile
  for (int s = 0; s < nsteps; s++) {
    const int k0 = s * BK;
    __syncthreads();
#pragma unroll
    for (int c = 0; c < 2; c++) {
      int chunk = ac0 + c;
      int row = chunk >> 2;
      int col = (chunk & 3) * 8;
      *(v8bf*)(&sA[0][row * LSTRIDE + col]) =
          *(const v8bf*)(Arow + (long)row * lda + k0 + col);
      *(v8bf*)(&sB[0][row * LSTRIDE + col]) =
          *(const v8bf*)(Brow + (long)row * ldb + k0 + col);
    }
    __syncthreads();
    if (s + 1 < nsteps) {
      __builtin_prefetch(Arow + (long)(ac0 >> 2) * lda + k0 + BK, 0, 1);
      __builtin_prefetch(Brow + (long)(ac0 >> 2) * ldb + k0 + BK, 0, 1);
    }
    mma_step(&sA[0][0], &sB[0][0], wm0, wn0, l16, lhalf, acc);
  }
#endif

  // ---- epilogue ----
  if (storeCT) {
    // C^T store: lane's 8 accum rows are contiguous -> two b128 stores/tile
#pragma unroll
    for (int mt = 0; mt < 4; mt++) {
#pragma unroll
      for (int nt = 0; nt < 2; nt++) {
        int gn = n0 + wn0 + nt * 16 + l16;
        int gm = m0 + wm0 + mt * 16 + lhalf * 8;
        float* p = C + (long)gn * ldc + gm;
        v4f lo, hi;
        lo[0] = acc[mt][nt][0]; lo[1] = acc[mt][nt][1];
        lo[2] = acc[mt][nt][2]; lo[3] = acc[mt][nt][3];
        hi[0] = acc[mt][nt][4]; hi[1] = acc[mt][nt][5];
        hi[2] = acc[mt][nt][6]; hi[3] = acc[mt][nt][7];
        *(v4f*)(p)     = lo;
        *(v4f*)(p + 4) = hi;
      }
    }
  } else {
#pragma unroll
    for (int mt = 0; mt < 4; mt++) {
#pragma unroll
      for (int nt = 0; nt < 2; nt++) {
        int gn = n0 + wn0 + nt * 16 + l16;
#pragma unroll
        for (int r = 0; r < 8; r++) {
          int gm = m0 + wm0 + mt * 16 + lhalf * 8 + r;
          C[(long)gm * ldc + gn] = acc[mt][nt][r];
        }
      }
    }
  }
}

// ---------------------------------------------------------------------------
// Orchestration: out[a] = x[a] @ W with W = sum_e w1[e] @ w2[e]
// (algebraic fusion: 550 GFLOP -> 77 GFLOP). All B operands pre-transposed so
// both GEMMs are NT with pure strided-2D tile DMA.
// ---------------------------------------------------------------------------
extern "C" void kernel_launch(void* const* d_in, const int* in_sizes, int n_in,
                              void* d_out, int out_size, void* d_ws, size_t ws_size,
                              hipStream_t stream) {
  const float* x  = (const float*)d_in[0];  // [4,1024,1024]
  const float* w1 = (const float*)d_in[1];  // [8,1024,4096]
  const float* w2 = (const float*)d_in[2];  // [8,4096,1024]
  float* out = (float*)d_out;               // [4,1024,1024] f32

  const long H = 1024, N = 4096, E = 8, AM = 4096;

  char* ws = (char*)d_ws;
  bf16_t* w1b    = (bf16_t*)ws;  ws += (size_t)E * H * N * sizeof(bf16_t);
  bf16_t* w2T    = (bf16_t*)ws;  ws += (size_t)E * H * N * sizeof(bf16_t);  // [E][H][N]
  bf16_t* xb     = (bf16_t*)ws;  ws += (size_t)AM * H * sizeof(bf16_t);
  float*  wpartT = (float*)ws;   ws += (size_t)E * H * H * sizeof(float);   // [E][h'][h]
  bf16_t* wcombT = (bf16_t*)ws;                                             // [h'][h]

  // 1) conversions (w2 transposed per expert: [N][H] f32 -> [H][N] bf16)
  cvt_f32_bf16<<<(unsigned)((E * H * N) / 2048), 256, 0, stream>>>(w1, w1b, E * H * N);
  {
    dim3 gt((unsigned)(H / 32), (unsigned)(N / 32), (unsigned)E);
    cvt_tr_f32_bf16<<<gt, 256, 0, stream>>>(w2, w2T, (int)N, (int)H, N * H, H * N);
  }
  cvt_f32_bf16<<<(unsigned)((AM * H) / 2048), 256, 0, stream>>>(x, xb, AM * H);

  // 2) wpartT[e] = (w1[e] @ w2[e])^T ; A = w1b[e] (HxN), B = w2T[e] (HxN, NT)
  dim3 g1((unsigned)(H / BN), (unsigned)(H / BM), (unsigned)E);  // 512 blocks
  gemm_bf16_wmma<<<g1, 256, 0, stream>>>(w1b, w2T, wpartT,
                                         (int)N, (int)N, (int)H, (int)N,
                                         H * N, H * N, H * H, /*storeCT=*/1);

  // 3) ordered expert sum (elementwise in transposed layout) -> bf16 W^T
  reduce8_bf16<<<(unsigned)((H * H) / 256), 256, 0, stream>>>(
      wpartT, wcombT, (int)(H * H), H * H);

  // 4) out = x @ W ; A = xb (AMxH), B = W^T (HxH, NT), normal C store
  dim3 g2((unsigned)(H / BN), (unsigned)(AM / BM), 1);  // 256 blocks
  gemm_bf16_wmma<<<g2, 256, 0, stream>>>(xb, wcombT, out,
                                         (int)H, (int)H, (int)H, (int)H,
                                         0, 0, 0, /*storeCT=*/0);
}
